// NTXentLoss_72507637891433
// MI455X (gfx1250) — matrix operand order repaired and checked by reading, not compiled, not run on hardware
//
#include <hip/hip_runtime.h>
#include <hip/hip_bf16.h>

typedef __attribute__((ext_vector_type(16))) _Float16 v16h;
typedef __attribute__((ext_vector_type(8)))  _Float16 v8h;
typedef __attribute__((ext_vector_type(4)))  _Float16 v4h;
typedef __attribute__((ext_vector_type(8)))  float    v8f;

#define DIM 128
#define INV_T (1.0f / 0.07f)

// ---------------------------------------------------------------------------
// Kernel 1: L2-normalize rows of [z_i; z_j] into f16 reps, and zero d_out.
// ---------------------------------------------------------------------------
__global__ __launch_bounds__(256)
void ntxent_normalize_kernel(const float* __restrict__ zi,
                             const float* __restrict__ zj,
                             _Float16* __restrict__ reps,
                             int Bh, float* __restrict__ out) {
    if (blockIdx.x == 0 && threadIdx.x == 0) *out = 0.0f;

    const int wave = threadIdx.x >> 5;
    const int lane = threadIdx.x & 31;
    const int row  = blockIdx.x * 8 + wave;

    const float* src = (row < Bh) ? (zi + (size_t)row * DIM)
                                  : (zj + (size_t)(row - Bh) * DIM);
    float4 x = *(const float4*)(src + lane * 4);
    float ss = x.x * x.x + x.y * x.y + x.z * x.z + x.w * x.w;
#pragma unroll
    for (int off = 16; off > 0; off >>= 1)
        ss += __shfl_xor(ss, off, 32);

    const float inv = 1.0f / fmaxf(sqrtf(ss), 1e-12f);
    v4h h;
    h[0] = (_Float16)(x.x * inv);
    h[1] = (_Float16)(x.y * inv);
    h[2] = (_Float16)(x.z * inv);
    h[3] = (_Float16)(x.w * inv);
    *(v4h*)(reps + (size_t)row * DIM + lane * 4) = h;
}

// ---------------------------------------------------------------------------
// Kernel 2: fused sim + exp + online logsumexp sweep (branch-free inner loop,
// double-buffered B tiles, one atomic per block).
// ---------------------------------------------------------------------------
__global__ __launch_bounds__(256)
void ntxent_main_kernel(const _Float16* __restrict__ reps,
                        float* __restrict__ out,
                        int N, int Bh) {
    const int tid  = threadIdx.x;
    const int lane = tid & 31;
    const int wave = tid >> 5;
    const int n    = lane & 15;      // column-within-tile (B/C/D layout)
    const int g    = lane >> 4;      // half-wave group
    const int rowBase = blockIdx.x * 16;
    const float invN  = 1.0f / (float)N;

    // ---- A fragments (16x32 f16 per K-chunk):
    //      lane L: M = L%16; halves i -> K = g*16+i
    const _Float16* arp = reps + (size_t)(rowBase + n) * DIM;
    v16h a[4];
#pragma unroll
    for (int kc = 0; kc < 4; ++kc) {
        const int b0 = kc * 32 + g * 16;
        v8h lo = *(const v8h*)(arp + b0);
        v8h hi = *(const v8h*)(arp + b0 + 8);
#pragma unroll
        for (int i = 0; i < 8; ++i) { a[kc][i] = lo[i]; a[kc][i + 8] = hi[i]; }
    }

    // Online logsumexp state + positive-logit accumulator.
    float m[8], s[8], pacc[8];
    int rowv[8], posv[8];
#pragma unroll
    for (int v = 0; v < 8; ++v) {
        m[v] = -1e30f; s[v] = 0.0f; pacc[v] = 0.0f;
        rowv[v] = rowBase + g * 8 + v;
        posv[v] = (rowv[v] < Bh) ? rowv[v] + Bh : rowv[v] - Bh;
    }

    const int ntiles = N / 16;
    const int tpw    = ntiles / 8;          // column tiles per wave (even)
    const int ct0    = wave * tpw;

    // B fragment loader: lane L: N = L%16; halves i -> K = kc*32 + g*16 + i
    auto loadB = [&](int ct, v16h (&bf)[4]) {
        const _Float16* brp = reps + (size_t)(ct * 16 + n) * DIM;
#pragma unroll
        for (int kc = 0; kc < 4; ++kc)
            bf[kc] = *(const v16h*)(brp + kc * 32 + g * 16);
    };

    auto computeTile = [&](int ct, v16h (&bf)[4]) {
        v8f acc = {};
#pragma unroll
        for (int kc = 0; kc < 4; ++kc)
            acc = __builtin_amdgcn_wmma_f32_16x16x32_f16(
                false, a[kc], false, bf[kc], (short)0, acc, false, false);

        const int col = ct * 16 + n;
#pragma unroll
        for (int v = 0; v < 8; ++v) {
            const float e = __expf(acc[v] * INV_T);       // e = exp(sim)
            pacc[v] += (col == posv[v]) ? e : 0.0f;       // positive logit
            const float em = (col == rowv[v]) ? -1e30f : e;  // mask diagonal
            const float mn = fmaxf(m[v], em);
            s[v] = s[v] * __expf(m[v] - mn) + __expf(em - mn);
            m[v] = mn;
        }
    };

    // ---- double-buffered sweep over this wave's column tiles
    v16h bf0[4], bf1[4];
    loadB(ct0, bf0);
    for (int t = 0; t < tpw; t += 2) {
        loadB(ct0 + t + 1, bf1);
        computeTile(ct0 + t, bf0);
        if (t + 2 < tpw) loadB(ct0 + t + 2, bf0);
        computeTile(ct0 + t + 1, bf1);
    }

    // ---- reduce (m,s,pacc) across the 16 lanes sharing each row set
#pragma unroll
    for (int off = 8; off > 0; off >>= 1) {
#pragma unroll
        for (int v = 0; v < 8; ++v) {
            const float mo = __shfl_xor(m[v], off, 32);
            const float so = __shfl_xor(s[v], off, 32);
            const float mn = fmaxf(m[v], mo);
            s[v] = s[v] * __expf(m[v] - mn) + so * __expf(mo - mn);
            m[v] = mn;
            pacc[v] += __shfl_xor(pacc[v], off, 32);
        }
    }

    // ---- combine the 8 waves' partials via LDS
    __shared__ float sm[8][16];
    __shared__ float ssum[8][16];
    __shared__ float spos[8][16];
    if (n == 0) {                       // lanes 0 and 16 hold reduced values
#pragma unroll
        for (int v = 0; v < 8; ++v) {
            sm[wave][g * 8 + v]   = m[v];
            ssum[wave][g * 8 + v] = s[v];
            spos[wave][g * 8 + v] = pacc[v];
        }
    }
    __syncthreads();

    if (tid < 16) {
        float M = sm[0][tid], S = ssum[0][tid], P = spos[0][tid];
#pragma unroll
        for (int w = 1; w < 8; ++w) {
            const float mo = sm[w][tid], so = ssum[w][tid];
            const float mn = fmaxf(M, mo);
            S = S * __expf(M - mn) + so * __expf(mo - mn);
            M = mn;
            P += spos[w][tid];
        }
        float loss = (M + __logf(S) - P) * invN;   // row (rowBase + tid)
        // sum the 16 per-row losses within wave 0, single atomic per block
#pragma unroll
        for (int off = 8; off > 0; off >>= 1)
            loss += __shfl_xor(loss, off, 32);
        if (tid == 0) atomicAdd(out, loss);
    }
}

// ---------------------------------------------------------------------------
extern "C" void kernel_launch(void* const* d_in, const int* in_sizes, int n_in,
                              void* d_out, int out_size, void* d_ws, size_t ws_size,
                              hipStream_t stream) {
    const float* zi = (const float*)d_in[0];
    const float* zj = (const float*)d_in[1];
    float* out      = (float*)d_out;

    const int Bh = in_sizes[0] / DIM;   // 8192
    const int N  = 2 * Bh;              // 16384

    _Float16* reps = (_Float16*)d_ws;   // N x 128 f16 = 4 MB (L2-resident)

    ntxent_normalize_kernel<<<N / 8, 256, 0, stream>>>(zi, zj, reps, Bh, out);
    ntxent_main_kernel<<<N / 16, 256, 0, stream>>>(reps, out, N, Bh);
}
